// PartWraper_64665027608892
// MI455X (gfx1250) — matrix operand order repaired and checked by reading, not compiled, not run on hardware
//
#include <hip/hip_runtime.h>
#include <hip/hip_bf16.h>

#define CIN   128
#define HH    376
#define WW    376
#define BB    4
#define NBOX  512
#define OUTC  28
#define TILEX 64
#define ROWS  4              // rows per workgroup -> 256 pixels
#define NPIX  (ROWS * TILEX)

typedef __bf16 bf16x16 __attribute__((ext_vector_type(16)));
typedef float  f32x8   __attribute__((ext_vector_type(8)));

union Frag { bf16x16 v; unsigned u[8]; };

// float -> bf16 (round to nearest even), pure bit math
__device__ __forceinline__ unsigned short f2bf(float f) {
    union { float f; unsigned u; } a; a.f = f;
    unsigned u = a.u;
    u = u + 0x7FFFu + ((u >> 16) & 1u);
    return (unsigned short)(u >> 16);
}

// B-fragment loader: N across lanes, K pairs per VGPR, +16 K-offset for hi lanes
__device__ __forceinline__ void load_b(Frag& f, const unsigned short* bp) {
    #pragma unroll
    for (int v = 0; v < 8; ++v) f.u[v] = *(const unsigned*)(bp + 2 * v);
}

// ---------------------------------------------------------------------------
// Kernel 0: repack w1 -> bf16 tap-major [9][o][i], w2 -> bf16 padded [32][128],
// fold BatchNorm into per-channel scale/shift.
// ---------------------------------------------------------------------------
__global__ __launch_bounds__(256) void prep_kernel(
    const float* __restrict__ w1, const float* __restrict__ gamma,
    const float* __restrict__ beta, const float* __restrict__ rmean,
    const float* __restrict__ rvar, const float* __restrict__ w2,
    unsigned short* __restrict__ w1b, unsigned short* __restrict__ w2b,
    float* __restrict__ scalep, float* __restrict__ shiftp)
{
    const int NW1 = 9 * CIN * CIN;
    int gid = blockIdx.x * 256 + threadIdx.x;
    if (gid < NW1) {
        int t = gid / (CIN * CIN);
        int rem = gid - t * (CIN * CIN);
        int o = rem >> 7, i = rem & 127;
        w1b[gid] = f2bf(w1[(o * CIN + i) * 9 + t]);           // OIHW, tap t
    } else if (gid < NW1 + 32 * CIN) {
        int e = gid - NW1;
        int o = e >> 7, k = e & 127;
        w2b[e] = (o < OUTC) ? f2bf(w2[o * CIN + k]) : (unsigned short)0;
    } else if (gid < NW1 + 32 * CIN + CIN) {
        int c = gid - NW1 - 32 * CIN;
        float s = gamma[c] * rsqrtf(rvar[c] + 1e-3f);
        scalep[c] = s;
        shiftp[c] = beta[c] - rmean[c] * s;
    }
}

// ---------------------------------------------------------------------------
// Kernel 1: fused 3x3 conv (implicit GEMM, bf16 WMMA) + BN + ReLU + 1x1 conv.
// One workgroup = 4 rows x 64 px = 256 pixels = 16 M-tiles. 8 waves:
//   conv1: wave w owns M-tiles {2w, 2w+1} x all 8 channel tiles (16 acc tiles)
//   kc-outer loop: all 9 taps of one 32-wide k-block stay resident in LDS;
//   B fragments ping-pong (loads for b[n+1] in flight while b[n] is consumed);
//   next k-block's weight slab prefetched (global_prefetch) during compute.
//   gemm2: wave w owns M-tiles {2w, 2w+1} x both out tiles (28 padded to 32)
// LDS: halo [6][66][136] + wslab [9*128][40] (mid aliases wslab) + w2 [32][136]
// ---------------------------------------------------------------------------
#define PAD   136
#define WPAD  40
#define SIN_ELEMS (6 * 66 * PAD)        // 53,856  (107,712 B)
#define SW_ELEMS  (9 * CIN * WPAD)      // 46,080  ( 92,160 B); mid = 34,816 fits
#define SW2_ELEMS (32 * PAD)            //  4,352  (  8,704 B)
#define SMEM_BYTES ((SIN_ELEMS + SW_ELEMS + SW2_ELEMS) * 2)   // 208,576 B

__global__ __launch_bounds__(256) void conv_fused_kernel(
    const float* __restrict__ feat,
    const unsigned short* __restrict__ w1b,
    const unsigned short* __restrict__ w2b,
    const float* __restrict__ scalep,
    const float* __restrict__ shiftp,
    float* __restrict__ confi)
{
    extern __shared__ char smem[];
    unsigned short* sIn  = (unsigned short*)smem;           // [6][66][PAD]
    unsigned short* sW   = sIn + SIN_ELEMS;                 // [9*128][WPAD]
    unsigned short* sMid = sW;                              // aliases sW (after conv1)
    unsigned short* sW2  = sW + SW_ELEMS;                   // [32][PAD]

    const int tid = threadIdx.x;
    const int x0  = blockIdx.x * TILEX;
    const int y0  = blockIdx.y * ROWS;
    const int b   = blockIdx.z;

    // ---- stage input halo: rows y0-1..y0+4, x0-1..x0+64, fp32 -> bf16 ----
    for (int idx = tid; idx < 6 * 66 * CIN; idx += 256) {
        int xi  = idx % 66;
        int rem = idx / 66;
        int row = rem % 6;
        int c   = rem / 6;
        int xx = x0 - 1 + xi;
        int yy = y0 - 1 + row;
        float v = 0.0f;
        if (xx >= 0 && xx < WW && yy >= 0 && yy < HH)
            v = feat[(((size_t)b * CIN + c) * HH + yy) * WW + xx];
        sIn[(row * 66 + xi) * PAD + c] = f2bf(v);
    }
    // ---- stage w2 (32x128 bf16 = 512 x uint4) ----
    {
        const uint4* src = (const uint4*)w2b;
        for (int it = tid; it < 512; it += 256) {
            int e = it * 8;
            int o = e >> 7, i = e & 127;
            *(uint4*)(sW2 + o * PAD + i) = src[it];
        }
    }

    const int wave = tid >> 5, lane = tid & 31;
    const int hi = lane >> 4, nl = lane & 15;
    const int mt0 = wave * 2;                 // first of the 2 M-tiles this wave owns
    const int pr0 = mt0 >> 2,      xt0 = (mt0 & 3) * 16;        // tile -> (row, xoff)
    const int pr1 = (mt0 + 1) >> 2, xt1 = ((mt0 + 1) & 3) * 16;

    f32x8 acc[2][8] = {};

    for (int kc = 0; kc < 4; ++kc) {
        __syncthreads();   // prev k-block's sW fully consumed (covers halo/w2 at kc=0)
        // stage weight slab: w1b[t][o][kc*32 .. kc*32+31] for all t,o
        // 9*128 rows x 32 bf16 = 4608 x uint4
        for (int it = tid; it < 4608; it += 256) {
            int row = it >> 2, q = it & 3;
            *(uint4*)(sW + row * WPAD + q * 8) =
                *(const uint4*)(w1b + row * CIN + kc * 32 + q * 8);
        }
        __syncthreads();

        // warm caches with next k-block's slab while this one is computed on
        if (kc < 3) {
            for (int it = tid; it < 9 * CIN; it += 256)
                __builtin_prefetch(w1b + it * CIN + (kc + 1) * 32, 0, 1);
        }

        const int kcOff = kc * 32;
        for (int dy = 0; dy < 3; ++dy) {
            for (int dx = 0; dx < 3; ++dx) {
                const int t = dy * 3 + dx;
                // A fragments for both M-tiles (contiguous K: k = 16*hi + s)
                Frag a0, a1;
                const unsigned short* ap0 =
                    sIn + ((pr0 + dy) * 66 + xt0 + nl + dx + 1) * PAD + kcOff + hi * 16;
                const unsigned short* ap1 =
                    sIn + ((pr1 + dy) * 66 + xt1 + nl + dx + 1) * PAD + kcOff + hi * 16;
                #pragma unroll
                for (int v = 0; v < 8; ++v) {
                    int k = 2 * v;
                    a0.u[v] = *(const unsigned*)(ap0 + k);
                    a1.u[v] = *(const unsigned*)(ap1 + k);
                }
                // B ping-pong: loads for the next fragment stay in flight while
                // the current one feeds two WMMAs (wait dscnt<=4 instead of 0)
                const unsigned short* bBase = sW + (t * CIN + nl) * WPAD + hi * 16;
                Frag b0, b1;
                load_b(b0, bBase);
                #pragma unroll
                for (int n = 0; n < 8; n += 2) {
                    load_b(b1, bBase + (n + 1) * 16 * WPAD);
                    acc[0][n] = __builtin_amdgcn_wmma_f32_16x16x32_bf16(
                        false, a0.v, false, b0.v, (short)0, acc[0][n], false, false);
                    acc[1][n] = __builtin_amdgcn_wmma_f32_16x16x32_bf16(
                        false, a1.v, false, b0.v, (short)0, acc[1][n], false, false);
                    if (n + 2 < 8)
                        load_b(b0, bBase + (n + 2) * 16 * WPAD);
                    acc[0][n + 1] = __builtin_amdgcn_wmma_f32_16x16x32_bf16(
                        false, a0.v, false, b1.v, (short)0, acc[0][n + 1], false, false);
                    acc[1][n + 1] = __builtin_amdgcn_wmma_f32_16x16x32_bf16(
                        false, a1.v, false, b1.v, (short)0, acc[1][n + 1], false, false);
                }
            }
        }
    }

    __syncthreads();   // all waves done reading sW/sIn before sMid (aliases sW)

    // ---- BN + ReLU -> bf16 mid[256][128] in LDS ----
    #pragma unroll
    for (int mt = 0; mt < 2; ++mt) {
        #pragma unroll
        for (int n = 0; n < 8; ++n) {
            int ch = n * 16 + nl;
            float sc = scalep[ch], sh = shiftp[ch];
            #pragma unroll
            for (int r = 0; r < 8; ++r) {
                float v = acc[mt][n][r] * sc + sh;    // C-layout: M = r + 8*hi
                v = v > 0.0f ? v : 0.0f;
                int p = (mt0 + mt) * 16 + r + hi * 8;
                sMid[p * PAD + ch] = f2bf(v);
            }
        }
    }
    __syncthreads();

    // ---- GEMM2: mid[256x128] x w2^T[128x32] -> confi tiles ----
    f32x8 acc2[2][2] = {};
    #pragma unroll
    for (int kc = 0; kc < 4; ++kc) {
        Frag a0, a1;
        const unsigned short* ap0 = sMid + (mt0 * 16 + nl) * PAD + kc * 32 + hi * 16;
        const unsigned short* ap1 = sMid + ((mt0 + 1) * 16 + nl) * PAD + kc * 32 + hi * 16;
        #pragma unroll
        for (int v = 0; v < 8; ++v) {
            int k = 2 * v;
            a0.u[v] = *(const unsigned*)(ap0 + k);
            a1.u[v] = *(const unsigned*)(ap1 + k);
        }
        #pragma unroll
        for (int n2 = 0; n2 < 2; ++n2) {
            Frag bf;
            load_b(bf, sW2 + (n2 * 16 + nl) * PAD + kc * 32 + hi * 16);
            acc2[0][n2] = __builtin_amdgcn_wmma_f32_16x16x32_bf16(
                false, a0.v, false, bf.v, (short)0, acc2[0][n2], false, false);
            acc2[1][n2] = __builtin_amdgcn_wmma_f32_16x16x32_bf16(
                false, a1.v, false, bf.v, (short)0, acc2[1][n2], false, false);
        }
    }
    #pragma unroll
    for (int mt = 0; mt < 2; ++mt) {
        #pragma unroll
        for (int n2 = 0; n2 < 2; ++n2) {
            int o = n2 * 16 + nl;
            if (o < OUTC) {
                #pragma unroll
                for (int r = 0; r < 8; ++r) {
                    int p  = (mt0 + mt) * 16 + r + hi * 8;
                    int yy = y0 + (p >> 6);
                    int xx = x0 + (p & 63);
                    if (xx < WW)
                        confi[(((size_t)b * OUTC + o) * HH + yy) * WW + xx] =
                            acc2[mt][n2][r];
                }
            }
        }
    }
}

// ---------------------------------------------------------------------------
// Kernel 2: rotated-grid bilinear sampling + mean + sigmoid. One wave per box.
// ---------------------------------------------------------------------------
__global__ __launch_bounds__(256) void sample_kernel(
    const float* __restrict__ confi,
    const float* __restrict__ boxes,
    float* __restrict__ out)
{
    const int tid = threadIdx.x;
    const int wave = tid >> 5, lane = tid & 31;
    const int box = blockIdx.x * 8 + wave;       // 0..2047
    const int b = box >> 9, n = box & (NBOX - 1);

    const float* bx = boxes + ((size_t)b * NBOX + n) * 5;
    float xg = bx[0], yg = bx[1], wg = bx[2], lg = bx[3], rg = bx[4];
    float cs = cosf(rg), sn = sinf(rg);

    float val = 0.0f;
    if (lane < OUTC) {
        int i = lane / 7, j = lane % 7;          // k = i*WIN_W + j
        float xx = ((float)i * (1.0f / 3.0f) - 0.5f) * wg;
        float yy = ((float)j * (1.0f / 6.0f) - 0.5f) * lg;
        float px = xx * cs + yy * sn + xg;
        float py = yy * cs - xx * sn + yg;
        float fx = px * ((float)WW / (WW - 1.0f)) - 0.5f;
        float fy = py * ((float)HH / (HH - 1.0f)) - 0.5f;
        float x0f = floorf(fx), y0f = floorf(fy);
        float wx = fx - x0f, wy = fy - y0f;
        const float* ch = confi + ((size_t)b * OUTC + lane) * (HH * WW);

        float v00 = 0.f, v10 = 0.f, v01 = 0.f, v11 = 0.f;
        {
            float xf = x0f, yf = y0f;
            if (xf >= 0.f && xf <= WW - 1.f && yf >= 0.f && yf <= HH - 1.f)
                v00 = ch[(int)yf * WW + (int)xf];
            xf = x0f + 1.f;
            if (xf >= 0.f && xf <= WW - 1.f && yf >= 0.f && yf <= HH - 1.f)
                v10 = ch[(int)yf * WW + (int)xf];
            xf = x0f; yf = y0f + 1.f;
            if (xf >= 0.f && xf <= WW - 1.f && yf >= 0.f && yf <= HH - 1.f)
                v01 = ch[(int)yf * WW + (int)xf];
            xf = x0f + 1.f;
            if (xf >= 0.f && xf <= WW - 1.f && yf >= 0.f && yf <= HH - 1.f)
                v11 = ch[(int)yf * WW + (int)xf];
        }
        val = v00 * (1.f - wx) * (1.f - wy) + v10 * wx * (1.f - wy)
            + v01 * (1.f - wx) * wy + v11 * wx * wy;
    }
    #pragma unroll
    for (int off = 16; off > 0; off >>= 1)
        val += __shfl_down(val, off, 32);
    if (lane == 0)
        out[box] = 1.0f / (1.0f + expf(-val * (1.0f / (float)OUTC)));
}

// ---------------------------------------------------------------------------
extern "C" void kernel_launch(void* const* d_in, const int* in_sizes, int n_in,
                              void* d_out, int out_size, void* d_ws, size_t ws_size,
                              hipStream_t stream) {
    const float* feat  = (const float*)d_in[0];
    const float* w1    = (const float*)d_in[1];
    const float* gamma = (const float*)d_in[2];
    const float* beta  = (const float*)d_in[3];
    const float* rmean = (const float*)d_in[4];
    const float* rvar  = (const float*)d_in[5];
    const float* w2    = (const float*)d_in[6];
    const float* boxes = (const float*)d_in[7];
    float* out = (float*)d_out;

    // workspace layout
    const size_t confiBytes = (size_t)BB * OUTC * HH * WW * sizeof(float); // 63,336,448
    const size_t w1bBytes   = (size_t)9 * CIN * CIN * 2;                   // 294,912
    const size_t w2bBytes   = (size_t)32 * CIN * 2;                        // 8,192
    char* ws = (char*)d_ws;
    float*          confi = (float*)ws;
    unsigned short* w1b   = (unsigned short*)(ws + confiBytes);
    unsigned short* w2b   = (unsigned short*)(ws + confiBytes + w1bBytes);
    float*          scp   = (float*)(ws + confiBytes + w1bBytes + w2bBytes);
    float*          shp   = scp + CIN;

    // 1) repack weights + fold BN
    int prepTasks = 9 * CIN * CIN + 32 * CIN + CIN;
    prep_kernel<<<(prepTasks + 255) / 256, 256, 0, stream>>>(
        w1, gamma, beta, rmean, rvar, w2, w1b, w2b, scp, shp);

    // 2) fused conv3x3 + BN + ReLU + conv1x1 (bf16 WMMA), 4-row tiles
    dim3 grid(WW / TILEX + (WW % TILEX ? 1 : 0), HH / ROWS, BB);   // 6 x 94 x 4
    conv_fused_kernel<<<grid, 256, SMEM_BYTES, stream>>>(
        feat, w1b, w2b, scp, shp, confi);

    // 3) rotated-grid bilinear sampling + sigmoid
    sample_kernel<<<(BB * NBOX) / 8, 256, 0, stream>>>(confi, boxes, out);
}
